// MHMOGATLayer_74182675136828
// MI455X (gfx1250) — compile-verified, matrix operand order checked by reading, not run on hardware
//
#include <hip/hip_runtime.h>

#define Nn 100000
#define Ee 1600000
#define IN_DIM 256
#define HCOLS 512          // [h1(84) h2(84) h3(84) self(256) pad(4)]
#define OUT_DIM 256
#define CDIM 508
#define NEG_SLOPE 0.2f
#define EPS 1e-5f

typedef __attribute__((ext_vector_type(16))) __bf16 v16bf;
typedef __attribute__((ext_vector_type(8)))  float  v8f;

union Frag16 { v16bf v; unsigned short u[16]; uint4 q[2]; };

__device__ __forceinline__ unsigned short f2bf(float f) {
    unsigned int u = __float_as_uint(f);
    unsigned int r = u + 0x7fffu + ((u >> 16) & 1u);   // RNE
    return (unsigned short)(r >> 16);
}
__device__ __forceinline__ float bf2f(unsigned short h) {
    return __uint_as_float(((unsigned int)h) << 16);
}
__device__ __forceinline__ unsigned int fkey(float f) {
    unsigned int u = __float_as_uint(f);
    return (u & 0x80000000u) ? ~u : (u | 0x80000000u);
}
__device__ __forceinline__ float fkey_dec(unsigned int k) {
    unsigned int u = (k & 0x80000000u) ? (k & 0x7fffffffu) : ~k;
    return __uint_as_float(u);
}

// ---------------- weight packing into exact B-fragment layout ----------------
// Bcat: 32 ntiles x 8 ksteps x 32 lanes x 16 bf16  (K=256, Ncols=512)
// Bproj: 16 ntiles x 16 ksteps x 32 lanes x 16 bf16 (K=512, Ncols=256)
__global__ __launch_bounds__(256) void k_pack(
    const float* __restrict__ W1, const float* __restrict__ W2,
    const float* __restrict__ W3, const float* __restrict__ Wself,
    const float* __restrict__ Wproj,
    unsigned short* __restrict__ Bcat, unsigned short* __restrict__ Bproj)
{
    int id = blockIdx.x * 256 + threadIdx.x;
    if (id < 32 * 8 * 32) {
        int nt = id >> 8, ks = (id >> 5) & 7, l = id & 31;
        int n = nt * 16 + (l & 15), half = l >> 4;
        unsigned short* dst = Bcat + (size_t)id * 16;
#pragma unroll
        for (int e = 0; e < 16; e++) {
            int K = ks * 32 + (e < 8 ? e : e + 8) + 8 * half;
            float v;
            if (n < 84)       v = W1[K * 84 + n];
            else if (n < 168) v = W2[K * 84 + (n - 84)];
            else if (n < 252) v = W3[K * 84 + (n - 168)];
            else if (n < 508) v = Wself[K * 256 + (n - 252)];
            else              v = 0.f;
            dst[e] = f2bf(v);
        }
    } else if (id < 32 * 8 * 32 + 16 * 16 * 32) {
        int id2 = id - 32 * 8 * 32;
        int nt = id2 >> 9, ks = (id2 >> 5) & 15, l = id2 & 31;
        int n = nt * 16 + (l & 15), half = l >> 4;
        unsigned short* dst = Bproj + (size_t)id2 * 16;
#pragma unroll
        for (int e = 0; e < 16; e++) {
            int K = ks * 32 + (e < 8 ? e : e + 8) + 8 * half;
            dst[e] = f2bf(K < CDIM ? Wproj[K * 256 + n] : 0.f);
        }
    }
}

// ---------------- GEMM1: H[N,512]bf16 = bf16(x) @ Bcat (+b_self on cols 252..507) --
__global__ __launch_bounds__(256) void k_gemm_cat(
    const float* __restrict__ x, const unsigned short* __restrict__ Bcat,
    const float* __restrict__ b_self, unsigned short* __restrict__ H)
{
    __shared__ unsigned short smem[16][264];   // 16 rows x 256 bf16, +8 pad
    int mtile = blockIdx.x, t = threadIdx.x;
    const float4* xb = (const float4*)(x + (size_t)mtile * 16 * IN_DIM);
#pragma unroll
    for (int k = 0; k < 4; k++) {
        int idx = t + k * 256;            // 1024 float4 = 16x256 floats
        int f = idx * 4, row = f >> 8, col = f & 255;
        float4 v = xb[idx];
        smem[row][col + 0] = f2bf(v.x);
        smem[row][col + 1] = f2bf(v.y);
        smem[row][col + 2] = f2bf(v.z);
        smem[row][col + 3] = f2bf(v.w);
    }
    __syncthreads();

    int wv = t >> 5, l = t & 31;
    int ntile = blockIdx.y * 8 + wv;      // 0..31
    int lm = l & 15, half = l >> 4;
    v8f c = {};
    const uint4* Bq = (const uint4*)Bcat + (size_t)ntile * 8 * 64;
#pragma unroll
    for (int ks = 0; ks < 8; ks++) {
        Frag16 a, b;
        int K0 = ks * 32 + 8 * half;
        a.q[0] = *(const uint4*)&smem[lm][K0];
        a.q[1] = *(const uint4*)&smem[lm][K0 + 16];
        b.q[0] = Bq[(size_t)ks * 64 + l * 2 + 0];
        b.q[1] = Bq[(size_t)ks * 64 + l * 2 + 1];
        c = __builtin_amdgcn_wmma_f32_16x16x32_bf16(false, a.v, false, b.v,
                                                    (short)0, c, false, false);
    }
    int n = ntile * 16 + lm;
    float bias = (n >= 252 && n < 508) ? b_self[n - 252] : 0.f;
    unsigned short* Hb = H + (size_t)(mtile * 16) * HCOLS + n;
#pragma unroll
    for (int r = 0; r < 8; r++) {
        int m = r + 8 * half;
        Hb[(size_t)m * HCOLS] = f2bf(c[r] + bias);
    }
}

// ---------------- attention logits per node/graph ----------------
__global__ __launch_bounds__(256) void k_att(
    const unsigned short* __restrict__ H,
    const float* __restrict__ as1, const float* __restrict__ ad1,
    const float* __restrict__ as2, const float* __restrict__ ad2,
    const float* __restrict__ as3, const float* __restrict__ ad3,
    float* __restrict__ ASRC, float* __restrict__ ADST)
{
    int n = blockIdx.x * 256 + threadIdx.x;
    int g = blockIdx.y;
    if (n >= Nn) return;
    const float* as = g == 0 ? as1 : (g == 1 ? as2 : as3);
    const float* ad = g == 0 ? ad1 : (g == 1 ? ad2 : ad3);
    const unsigned short* hr = H + (size_t)n * HCOLS + g * 84;
    float s0 = 0.f, s1 = 0.f, d0 = 0.f, d1 = 0.f;
#pragma unroll 7
    for (int d = 0; d < 42; d++) {
        float f = bf2f(hr[d]);
        s0 += f * as[d];  d0 += f * ad[d];
    }
#pragma unroll 7
    for (int d = 0; d < 42; d++) {
        float f = bf2f(hr[42 + d]);
        s1 += f * as[42 + d];  d1 += f * ad[42 + d];
    }
    size_t o = ((size_t)g * Nn + n) * 2;
    ASRC[o] = s0; ASRC[o + 1] = s1;
    ADST[o] = d0; ADST[o + 1] = d1;
}

// ---------------- zero scratch (AGG | DEN | AMAXK contiguous) ----------------
__global__ __launch_bounds__(256) void k_zero(uint4* __restrict__ p, int count)
{
    int i = blockIdx.x * 256 + threadIdx.x;
    if (i < count) { uint4 z = {0, 0, 0, 0}; p[i] = z; }
}

// ---------------- edge pass 1: segment max (monotone-key atomicMax) ----------
__global__ __launch_bounds__(256) void k_edge_max(
    const int* __restrict__ ei1, const int* __restrict__ ei2, const int* __restrict__ ei3,
    const float* __restrict__ ASRC, const float* __restrict__ ADST,
    unsigned int* __restrict__ AMAXK)
{
    int e = blockIdx.x * 256 + threadIdx.x;
    int g = blockIdx.y;
    if (e >= Ee + Nn) return;
    const int* ei = g == 0 ? ei1 : (g == 1 ? ei2 : ei3);
    int s, d;
    if (e < Ee) { s = ei[e]; d = ei[Ee + e]; } else { s = d = e - Ee; }
    size_t gb = (size_t)g * Nn * 2;
#pragma unroll
    for (int h = 0; h < 2; h++) {
        float a = ASRC[gb + (size_t)s * 2 + h] + ADST[gb + (size_t)d * 2 + h];
        a = a > 0.f ? a : NEG_SLOPE * a;
        atomicMax(&AMAXK[gb + (size_t)d * 2 + h], fkey(a));
    }
}

// ---------------- edge pass 2: segment sum of exp ----------------
__global__ __launch_bounds__(256) void k_edge_sum(
    const int* __restrict__ ei1, const int* __restrict__ ei2, const int* __restrict__ ei3,
    const float* __restrict__ ASRC, const float* __restrict__ ADST,
    const unsigned int* __restrict__ AMAXK, float* __restrict__ DEN)
{
    int e = blockIdx.x * 256 + threadIdx.x;
    int g = blockIdx.y;
    if (e >= Ee + Nn) return;
    const int* ei = g == 0 ? ei1 : (g == 1 ? ei2 : ei3);
    int s, d;
    if (e < Ee) { s = ei[e]; d = ei[Ee + e]; } else { s = d = e - Ee; }
    size_t gb = (size_t)g * Nn * 2;
#pragma unroll
    for (int h = 0; h < 2; h++) {
        float a = ASRC[gb + (size_t)s * 2 + h] + ADST[gb + (size_t)d * 2 + h];
        a = a > 0.f ? a : NEG_SLOPE * a;
        float amax = fkey_dec(AMAXK[gb + (size_t)d * 2 + h]);
        atomicAdd(&DEN[gb + (size_t)d * 2 + h], __expf(a - amax));
    }
}

// ---------------- edge pass 3: weighted message aggregation ----------------
__global__ __launch_bounds__(256) void k_edge_msg(
    const int* __restrict__ ei1, const int* __restrict__ ei2, const int* __restrict__ ei3,
    const float* __restrict__ ASRC, const float* __restrict__ ADST,
    const unsigned int* __restrict__ AMAXK, const float* __restrict__ DEN,
    const unsigned short* __restrict__ H, float* __restrict__ AGG)
{
    int e = blockIdx.x * 256 + threadIdx.x;
    int g = blockIdx.y;
    if (e >= Ee + Nn) return;
    const int* ei = g == 0 ? ei1 : (g == 1 ? ei2 : ei3);
    int s, d;
    if (e < Ee) { s = ei[e]; d = ei[Ee + e]; } else { s = d = e - Ee; }
    size_t gb = (size_t)g * Nn * 2;
    float w[2];
#pragma unroll
    for (int h = 0; h < 2; h++) {
        float a = ASRC[gb + (size_t)s * 2 + h] + ADST[gb + (size_t)d * 2 + h];
        a = a > 0.f ? a : NEG_SLOPE * a;
        float amax = fkey_dec(AMAXK[gb + (size_t)d * 2 + h]);
        w[h] = __expf(a - amax) / DEN[gb + (size_t)d * 2 + h];
    }
    const uint2* hr = (const uint2*)(H + (size_t)s * HCOLS + g * 84);  // 8B aligned
    float* ar = AGG + (size_t)d * 256 + g * 84;
#pragma unroll 3
    for (int ch = 0; ch < 21; ch++) {
        uint2 p = hr[ch];
        int db = ch * 4;
        float f0 = bf2f((unsigned short)(p.x & 0xffffu));
        float f1 = bf2f((unsigned short)(p.x >> 16));
        float f2 = bf2f((unsigned short)(p.y & 0xffffu));
        float f3 = bf2f((unsigned short)(p.y >> 16));
        atomicAdd(ar + db + 0, f0 * w[(db + 0) < 42 ? 0 : 1]);
        atomicAdd(ar + db + 1, f1 * w[(db + 1) < 42 ? 0 : 1]);
        atomicAdd(ar + db + 2, f2 * w[(db + 2) < 42 ? 0 : 1]);
        atomicAdd(ar + db + 3, f3 * w[(db + 3) < 42 ? 0 : 1]);
    }
}

// ---------------- build bf16 concat matrix [N,512] ----------------
__global__ __launch_bounds__(256) void k_finalize(
    const float* __restrict__ AGG, const unsigned short* __restrict__ H,
    const float* __restrict__ b1, const float* __restrict__ b2, const float* __restrict__ b3,
    unsigned short* __restrict__ CONC)
{
    size_t idx = (size_t)blockIdx.x * 256 + threadIdx.x;
    int n = (int)(idx >> 9), c = (int)(idx & 511);
    unsigned short v;
    if (c < 252) {
        int g = c / 84, cc = c - g * 84;
        const float* bias = g == 0 ? b1 : (g == 1 ? b2 : b3);
        v = f2bf(AGG[(size_t)n * 256 + c] + bias[cc]);
    } else if (c < 508) {
        v = H[(size_t)n * HCOLS + c];
    } else v = 0;
    CONC[idx] = v;
}

// ---------------- GEMM2: out[N,256]f32 = CONC(bf16) @ Bproj + b_proj ----------
__global__ __launch_bounds__(256) void k_gemm_proj(
    const unsigned short* __restrict__ CONC, const unsigned short* __restrict__ Bproj,
    const float* __restrict__ b_proj, float* __restrict__ out)
{
    __shared__ unsigned short smem[16][520];   // 16 rows x 512 bf16, +8 pad
    int mtile = blockIdx.x, t = threadIdx.x;
    const uint4* cb = (const uint4*)(CONC + (size_t)mtile * 16 * HCOLS);
#pragma unroll
    for (int k = 0; k < 4; k++) {
        int idx = t + k * 256;            // 1024 uint4 = 16x512 bf16
        int f = idx * 8, row = f >> 9, col = f & 511;
        *(uint4*)&smem[row][col] = cb[idx];
    }
    __syncthreads();

    int wv = t >> 5, l = t & 31;
    int ntile = blockIdx.y * 8 + wv;      // 0..15
    int lm = l & 15, half = l >> 4;
    v8f c = {};
    const uint4* Bq = (const uint4*)Bproj + (size_t)ntile * 16 * 64;
#pragma unroll
    for (int ks = 0; ks < 16; ks++) {
        Frag16 a, b;
        int K0 = ks * 32 + 8 * half;
        a.q[0] = *(const uint4*)&smem[lm][K0];
        a.q[1] = *(const uint4*)&smem[lm][K0 + 16];
        b.q[0] = Bq[(size_t)ks * 64 + l * 2 + 0];
        b.q[1] = Bq[(size_t)ks * 64 + l * 2 + 1];
        c = __builtin_amdgcn_wmma_f32_16x16x32_bf16(false, a.v, false, b.v,
                                                    (short)0, c, false, false);
    }
    int n = ntile * 16 + lm;
    float bias = b_proj[n];
    float* ob = out + (size_t)(mtile * 16) * OUT_DIM + n;
#pragma unroll
    for (int r = 0; r < 8; r++) {
        int m = r + 8 * half;
        ob[(size_t)m * OUT_DIM] = c[r] + bias;
    }
}

// ---------------- in-place layernorm, one wave32 per row ----------------
__global__ __launch_bounds__(256) void k_ln(
    float* __restrict__ out, const float* __restrict__ gamma, const float* __restrict__ beta)
{
    int wv = threadIdx.x >> 5, l = threadIdx.x & 31;
    int n = blockIdx.x * 8 + wv;
    if (n >= Nn) return;
    float* row = out + (size_t)n * OUT_DIM;
    float v[8], s = 0.f;
#pragma unroll
    for (int k = 0; k < 8; k++) { v[k] = row[l + 32 * k]; s += v[k]; }
#pragma unroll
    for (int off = 16; off > 0; off >>= 1) s += __shfl_xor(s, off, 32);
    float mu = s * (1.f / 256.f);
    float q = 0.f;
#pragma unroll
    for (int k = 0; k < 8; k++) { float dd = v[k] - mu; q += dd * dd; }
#pragma unroll
    for (int off = 16; off > 0; off >>= 1) q += __shfl_xor(q, off, 32);
    float inv = rsqrtf(q * (1.f / 256.f) + EPS);
#pragma unroll
    for (int k = 0; k < 8; k++) {
        int cc = l + 32 * k;
        row[cc] = (v[k] - mu) * inv * gamma[cc] + beta[cc];
    }
}

extern "C" void kernel_launch(void* const* d_in, const int* in_sizes, int n_in,
                              void* d_out, int out_size, void* d_ws, size_t ws_size,
                              hipStream_t stream) {
    const float* x     = (const float*)d_in[0];
    const int*   ei1   = (const int*)d_in[1];
    const float* W1    = (const float*)d_in[2];
    const float* as1   = (const float*)d_in[3];
    const float* ad1   = (const float*)d_in[4];
    const float* b1    = (const float*)d_in[5];
    const int*   ei2   = (const int*)d_in[6];
    const float* W2    = (const float*)d_in[7];
    const float* as2   = (const float*)d_in[8];
    const float* ad2   = (const float*)d_in[9];
    const float* b2    = (const float*)d_in[10];
    const int*   ei3   = (const int*)d_in[11];
    const float* W3    = (const float*)d_in[12];
    const float* as3   = (const float*)d_in[13];
    const float* ad3   = (const float*)d_in[14];
    const float* b3    = (const float*)d_in[15];
    const float* Wself = (const float*)d_in[16];
    const float* bself = (const float*)d_in[17];
    const float* Wproj = (const float*)d_in[18];
    const float* bproj = (const float*)d_in[19];
    const float* gamma = (const float*)d_in[20];
    const float* beta  = (const float*)d_in[21];
    float* out = (float*)d_out;

    char* ws = (char*)d_ws;
    unsigned short* H     = (unsigned short*)(ws);                 // N*512*2   = 102,400,000
    unsigned short* CONC  = (unsigned short*)(ws + 102400000ull);  // N*512*2
    float*          AGG   = (float*)(ws + 204800000ull);           // N*256*4   = 102,400,000
    float*          DEN   = (float*)(ws + 307200000ull);           // 3*N*2*4   =   2,400,000
    unsigned int*   AMAXK = (unsigned int*)(ws + 309600000ull);    // 3*N*2*4
    float*          ASRC  = (float*)(ws + 312000000ull);           // 3*N*2*4
    float*          ADST  = (float*)(ws + 314400000ull);           // 3*N*2*4
    unsigned short* Bcat  = (unsigned short*)(ws + 316800000ull);  // 262,144
    unsigned short* Bproj = (unsigned short*)(ws + 317062144ull);  // 262,144

    // 1) pack weights (bf16, exact WMMA B-fragment layout)
    k_pack<<<64, 256, 0, stream>>>(W1, W2, W3, Wself, Wproj, Bcat, Bproj);

    // 2) H = bf16( x @ [W1|W2|W3|W_self] ), +b_self on self cols
    k_gemm_cat<<<dim3(Nn / 16, 4), 256, 0, stream>>>(x, Bcat, bself, H);

    // 3) per-node attention logits
    k_att<<<dim3((Nn + 255) / 256, 3), 256, 0, stream>>>(H, as1, ad1, as2, ad2, as3, ad3, ASRC, ADST);

    // 4) zero AGG|DEN|AMAXK (contiguous 107,200,000 bytes -> 6,700,000 uint4)
    {
        int cnt = 6700000;
        k_zero<<<(cnt + 255) / 256, 256, 0, stream>>>((uint4*)AGG, cnt);
    }

    // 5) 3-pass per-dst softmax + aggregation over edges (+self loops), all graphs
    dim3 egrid((Ee + Nn + 255) / 256, 3);
    k_edge_max<<<egrid, 256, 0, stream>>>(ei1, ei2, ei3, ASRC, ADST, AMAXK);
    k_edge_sum<<<egrid, 256, 0, stream>>>(ei1, ei2, ei3, ASRC, ADST, AMAXK, DEN);
    k_edge_msg<<<egrid, 256, 0, stream>>>(ei1, ei2, ei3, ASRC, ADST, AMAXK, DEN, H, AGG);

    // 6) build bf16 concat features [N,512]
    k_finalize<<<(Nn * 512) / 256, 256, 0, stream>>>(AGG, H, b1, b2, b3, CONC);

    // 7) projection GEMM + bias
    k_gemm_proj<<<dim3(Nn / 16, 2), 256, 0, stream>>>(CONC, Bproj, bproj, out);

    // 8) in-place layernorm
    k_ln<<<Nn / 8, 256, 0, stream>>>(out, gamma, beta);
}
